// ProcessNoiseNetwork_31817117729081
// MI455X (gfx1250) — matrix operand order repaired and checked by reading, not compiled, not run on hardware
//
#include <hip/hip_runtime.h>

typedef __attribute__((ext_vector_type(16))) _Float16 v16h;
typedef __attribute__((ext_vector_type(8)))  float    v8f;
typedef __attribute__((ext_vector_type(2)))  float    v2f;

#define SD   32
#define HID  64
#define OUTD 528            // 32*33/2 packed lower-tri
#define WAVES 4
#define TPB  (WAVES * 32)
#define ELEM_F 768          // per-element L storage: rows 0-15 stride 16, rows 16-31 stride 32

// LDS layout (bytes). Weights stored FRAGMENT-LINEAR: frag f = nt*2 + kstep,
// halfword index = f*512 + lane*16 + e  (one aligned 32B v16h per lane per frag).
#define SMEM_W1     0                   // 8 frags  * 1KB = 8192
#define SMEM_W2     8192                // 8 frags  * 1KB = 8192
#define SMEM_W3     16384               // 66 frags * 1KB = 67584
#define SMEM_WAVE   83968               // 16-aligned
#define WAVE_BYTES  51200               // hbuf 16x64 f16 (2048) + L 16x768 f32 (49152)
#define SMEM_TOTAL  (SMEM_WAVE + WAVES * WAVE_BYTES)   // 288768 B (~282 KB <= 320 KB)

__device__ __forceinline__ void lds_fence() {
  asm volatile("s_wait_dscnt 0" ::: "memory");
}

// Branch-free transcendentals on the hardware v_exp_f32 / v_rcp_f32 paths.
__device__ __forceinline__ float fast_exp(float x) {
  return __builtin_amdgcn_exp2f(x * 1.4426950408889634f);
}
__device__ __forceinline__ float fast_tanh(float x) {
  float e = __builtin_amdgcn_exp2f(x * 2.8853900817779268f);
  return 1.0f - 2.0f * __builtin_amdgcn_rcpf(e + 1.0f);
}

// A-operand K order used here (contiguous, same as B staging): lane group g (lane>>4), element e -> K index
__device__ __forceinline__ int akmap(int e, int g) {
  return e + 16 * g;
}

// B fragment: one contiguous 32B chunk per lane from fragment-linear storage.
__device__ __forceinline__ v16h bfrag_lin(const _Float16* Ws, int f, int lane) {
  return *(const v16h*)(Ws + (f << 9) + (lane << 4));
}

__device__ __forceinline__ v8f wmma_f16(v16h a, v16h b, v8f c) {
  return __builtin_amdgcn_wmma_f32_16x16x32_f16(false, a, false, b, (short)0, c, false, false);
}
// Full-precision f32 WMMA, 16x16 tile, K=4. A: lane m=lane&15, VGPR pair = K (k0+2g, k0+2g+1).
// B mirrors A with N on lanes (for Q = L*L^T both fragments hold identical lane data).
__device__ __forceinline__ v8f wmma_f32k4(v2f a, v2f b, v8f c) {
  return __builtin_amdgcn_wmma_f32_16x16x4_f32(false, a, false, b, (short)0, c, false, false);
}

__global__ __launch_bounds__(TPB, 1)
void ProcessNoiseNetwork_kernel(const float* __restrict__ t, const float* __restrict__ x,
                                const float* __restrict__ W1, const float* __restrict__ b1,
                                const float* __restrict__ W2, const float* __restrict__ b2,
                                const float* __restrict__ W3, const float* __restrict__ b3,
                                float* __restrict__ out, int B) {
  extern __shared__ char smem[];
  _Float16* W1s = (_Float16*)(smem + SMEM_W1);
  _Float16* W2s = (_Float16*)(smem + SMEM_W2);
  _Float16* W3s = (_Float16*)(smem + SMEM_W3);

  const int tid  = threadIdx.x;
  const int lane = tid & 31;
  const int wv   = tid >> 5;
  const int m    = lane & 15;   // tile row (A/M), also C column-lane
  const int grp  = lane >> 4;   // lane half

  _Float16* hbuf = (_Float16*)(smem + SMEM_WAVE + wv * WAVE_BYTES);
  float*    Lp   = (float*)(smem + SMEM_WAVE + wv * WAVE_BYTES + 2048);

  // ---- Stage weights into LDS as f16, pre-swizzled into B-fragment order ----
  for (int d = tid; d < HID * HID; d += TPB) {
    int f = d >> 9, L = (d >> 4) & 31, e = d & 15;
    int k = ((f & 1) << 5) + ((L >> 4) << 4) + e;
    int n = ((f >> 1) << 4) + (L & 15);
    W1s[d] = (_Float16)((k < SD + 1) ? W1[k * HID + n] : 0.f);
    W2s[d] = (_Float16)W2[k * HID + n];
  }
  for (int d = tid; d < HID * OUTD; d += TPB) {
    int f = d >> 9, L = (d >> 4) & 31, e = d & 15;
    int k = ((f & 1) << 5) + ((L >> 4) << 4) + e;
    int n = ((f >> 1) << 4) + (L & 15);
    W3s[d] = (_Float16)W3[k * OUTD + n];
  }
  __syncthreads();

  const int tile = blockIdx.x * WAVES + wv;
  const int rowBase = tile * 16;
  if (rowBase >= B) return;
  const int row = rowBase + m;

  // ---- Zero this wave's L buffer (upper-triangle K slots must read as 0) ----
  {
    float4 z4; z4.x = z4.y = z4.z = z4.w = 0.f;
    float4* lp4 = (float4*)Lp;
    for (int i = lane; i < 16 * (ELEM_F / 4); i += 32) lp4[i] = z4;
  }

  // ---- GEMM1: tx(16x64, zero-padded) @ W1(64x64) -> tanh -> hbuf ----
  v16h a0, a1;
#pragma unroll
  for (int e = 0; e < 16; ++e) {
    int k0 = akmap(e, grp);
    int k1 = k0 + 32;
    float v0 = (k0 == 0) ? t[row] : ((k0 < 33) ? x[(size_t)row * SD + (k0 - 1)] : 0.f);
    float v1 = (k1 < 33) ? x[(size_t)row * SD + (k1 - 1)] : 0.f;
    a0[e] = (_Float16)v0;
    a1[e] = (_Float16)v1;
  }
#pragma unroll
  for (int nt = 0; nt < 4; ++nt) {
    v8f c = {};
    c = wmma_f16(a0, bfrag_lin(W1s, nt * 2 + 0, lane), c);
    c = wmma_f16(a1, bfrag_lin(W1s, nt * 2 + 1, lane), c);
    int col = nt * 16 + m;
    float bb = b1[col];
#pragma unroll
    for (int r = 0; r < 8; ++r)
      hbuf[(r + 8 * grp) * HID + col] = (_Float16)fast_tanh(c[r] + bb);
  }
  lds_fence();

  // ---- GEMM2: h1(16x64) @ W2(64x64) -> tanh -> hbuf (in place after frag loads) ----
  v16h h0, h1;
#pragma unroll
  for (int e = 0; e < 16; ++e) {
    h0[e] = hbuf[m * HID + akmap(e, grp)];
    h1[e] = hbuf[m * HID + 32 + akmap(e, grp)];
  }
  lds_fence();
#pragma unroll
  for (int nt = 0; nt < 4; ++nt) {
    v8f c = {};
    c = wmma_f16(h0, bfrag_lin(W2s, nt * 2 + 0, lane), c);
    c = wmma_f16(h1, bfrag_lin(W2s, nt * 2 + 1, lane), c);
    int col = nt * 16 + m;
    float bb = b2[col];
#pragma unroll
    for (int r = 0; r < 8; ++r)
      hbuf[(r + 8 * grp) * HID + col] = (_Float16)fast_tanh(c[r] + bb);
  }
  lds_fence();

  // ---- GEMM3: h2(16x64) @ W3(64x528) + b3 -> dense-ish L rows, exp on diagonal ----
  v16h q0, q1;
#pragma unroll
  for (int e = 0; e < 16; ++e) {
    q0[e] = hbuf[m * HID + akmap(e, grp)];
    q1[e] = hbuf[m * HID + 32 + akmap(e, grp)];
  }
  lds_fence();
#pragma unroll 4
  for (int nt = 0; nt < 33; ++nt) {
    v8f c = {};
    c = wmma_f16(q0, bfrag_lin(W3s, nt * 2 + 0, lane), c);
    c = wmma_f16(q1, bfrag_lin(W3s, nt * 2 + 1, lane), c);
    int cn = nt * 16 + m;                         // packed tri index, fixed per lane
    int i  = (int)((sqrtf(8.f * (float)cn + 9.f) - 3.f) * 0.5f + 1e-4f);
    while ((i + 1) * (i + 2) / 2 <= cn) ++i;      // fp-safety clamp
    while (i * (i + 1) / 2 > cn) --i;
    int  kk   = cn - i * (i + 1) / 2;
    bool diag = (kk == i);
    // L row layout: rows 0-15 stride 16 @ i*16 ; rows 16-31 stride 32 @ 256+(i-16)*32
    int  dst  = (i < 16) ? (i * 16 + kk) : (256 + ((i - 16) << 5) + kk);
    float bb  = b3[cn];
#pragma unroll
    for (int r = 0; r < 8; ++r) {
      float v = c[r] + bb;
      float ve = fast_exp(v) + 1e-6f;
      Lp[(r + 8 * grp) * ELEM_F + dst] = diag ? ve : v;
    }
  }
  lds_fence();

  // ---- Q = L @ L^T per element via V_WMMA_F32_16X16X4_F32 ----
  // Fragment for tile tt, k-quad kq: lane holds L[tt*16 + (lane&15)][kq*4 + 2*grp .. +1]
  // (identical data serves as both A and B operand). C10 = C01^T stored transposed.
  const int rb0 = m * 16;                 // row m,     k base (rows 0-15, stride 16)
  const int rb1 = 256 + m * 32;           // row 16+m,  k base (rows 16-31, stride 32)
  const int ko  = 2 * grp;

#pragma unroll 2
  for (int e = 0; e < 16; ++e) {
    const float* Le = Lp + e * ELEM_F;
    v2f F0[4], F1[8];
#pragma unroll
    for (int kq = 0; kq < 4; ++kq) F0[kq] = *(const v2f*)(Le + rb0 + kq * 4 + ko);
#pragma unroll
    for (int kq = 0; kq < 8; ++kq) F1[kq] = *(const v2f*)(Le + rb1 + kq * 4 + ko);

    v8f c00 = {}, c01 = {}, c11 = {};
#pragma unroll
    for (int kq = 0; kq < 4; ++kq) {      // K <= 15 suffices for tiles touching rows/cols 0-15
      c00 = wmma_f32k4(F0[kq], F0[kq], c00);
      c01 = wmma_f32k4(F0[kq], F1[kq], c01);
    }
#pragma unroll
    for (int kq = 0; kq < 8; ++kq)
      c11 = wmma_f32k4(F1[kq], F1[kq], c11);

    float* oute = out + (size_t)(rowBase + e) * (size_t)(SD * SD);
#pragma unroll
    for (int r = 0; r < 8; ++r) {
      int i = r + 8 * grp;
      oute[i * SD + m]             = c00[r];   // Q[0:16, 0:16]
      oute[i * SD + 16 + m]        = c01[r];   // Q[0:16, 16:32]
      oute[(16 + i) * SD + 16 + m] = c11[r];   // Q[16:32,16:32]
    }
    // Q[16:32, 0:16] = C01^T : lane writes 8 contiguous floats of row (16+m)
    {
      float4 s0, s1;
      s0.x = c01[0]; s0.y = c01[1]; s0.z = c01[2]; s0.w = c01[3];
      s1.x = c01[4]; s1.y = c01[5]; s1.z = c01[6]; s1.w = c01[7];
      float* p = oute + (16 + m) * SD + 8 * grp;
      *(float4*)(p + 0) = s0;
      *(float4*)(p + 4) = s1;
    }
  }
}

extern "C" void kernel_launch(void* const* d_in, const int* in_sizes, int n_in,
                              void* d_out, int out_size, void* d_ws, size_t ws_size,
                              hipStream_t stream) {
  const float* t  = (const float*)d_in[0];
  const float* x  = (const float*)d_in[1];
  const float* W1 = (const float*)d_in[2];
  const float* b1 = (const float*)d_in[3];
  const float* W2 = (const float*)d_in[4];
  const float* b2 = (const float*)d_in[5];
  const float* W3 = (const float*)d_in[6];
  const float* b3 = (const float*)d_in[7];
  float* out = (float*)d_out;

  const int B = in_sizes[0];                     // 131072
  const int tiles = B / 16;                      // 8192 wave-tiles
  const int grid = (tiles + WAVES - 1) / WAVES;  // 2048 blocks of 128 threads (4 waves)
  ProcessNoiseNetwork_kernel<<<grid, TPB, SMEM_TOTAL, stream>>>(t, x, W1, b1, W2, b2, W3, b3, out, B);
}
